// MultiScaleDeformableAttention_21973052686739
// MI455X (gfx1250) — matrix operand order repaired and checked by reading, not compiled, not run on hardware
//
#include <hip/hip_runtime.h>

// ---------------------------------------------------------------------------
// MultiScaleDeformableAttention for MI455X (gfx1250, wave32, WMMA bf16)
//   B=4, Q=8192, D=256, NH=8, hd=32, NL=4, NP=4, feats 100/50/25/13 (square)
// Pipeline:
//   prep:    fp32 weights -> bf16, column-major [n][k] for WMMA B-operand
//   gemm1:   lin = query @ [w_off|w_attn] + bias     (N=384, WMMA bf16)
//   gemmval: val[b][pos][256] = feat^T @ w_val^T     (per level, WMMA bf16)
//   sampler: tanh + softmax + bilinear gather + weighted accumulate
//   gemm2:   out = accum @ w_out + b_out             (N=256, WMMA bf16)
// ---------------------------------------------------------------------------

#define B_     4
#define Q_     8192
#define NHEAD  8
#define TOT_HW 13294   // 10000 + 2500 + 625 + 169

typedef __attribute__((ext_vector_type(16))) __bf16 v16bf;
typedef __attribute__((ext_vector_type(8)))  float  v8f;

union BfPack { uint4 u[2]; v16bf v; };

__device__ __forceinline__ unsigned short f2bf(float f) {
  unsigned int u = __float_as_uint(f);
  u += 0x7FFFu + ((u >> 16) & 1u);          // round-to-nearest-even
  return (unsigned short)(u >> 16);
}

// ---------------------------------------------------------------------------
// Weight prep: build bf16 weights (column-major [n][k] for the B operand) and
// the concatenated fp32 bias for the fused offset/attn GEMM.
// ---------------------------------------------------------------------------
__global__ __launch_bounds__(256) void prep_kernel(
    const float* __restrict__ w_off,  const float* __restrict__ b_off,
    const float* __restrict__ w_attn, const float* __restrict__ b_attn,
    const float* __restrict__ w_val,  const float* __restrict__ w_out,
    unsigned short* __restrict__ wcat, unsigned short* __restrict__ wval,
    unsigned short* __restrict__ wout, float* __restrict__ bias_cat)
{
  const int i = blockIdx.x * 256 + threadIdx.x;
  if (i < 384 * 256) {                       // [n=384][k=256] <- w_off|w_attn ([k][n])
    const int n = i >> 8, k = i & 255;
    const float f = (n < 256) ? w_off[k * 256 + n] : w_attn[k * 128 + (n - 256)];
    wcat[i] = f2bf(f);
  }
  if (i < 4 * 256 * 256)                     // w_val[l][o][i] is already [n=o][k=i]
    wval[i] = f2bf(w_val[i]);
  if (i < 256 * 256) {                       // [n][k] <- w_out ([k][n])
    const int n = i >> 8, k = i & 255;
    wout[i] = f2bf(w_out[k * 256 + n]);
  }
  if (i < 384) bias_cat[i] = (i < 256) ? b_off[i] : b_attn[i - 256];
}

// ---------------------------------------------------------------------------
// Shared WMMA K-loop: A tile in LDS as bf16 [16 rows][256 K] (+8 pad halfs),
// B weights in global, column-major [n][256]. Each wave owns NT 16x16 tiles.
// A lane striping (16-bit A 16x32): lane m, 16 contiguous halfs at k0 + 16*(lane>=16).
// B lane striping: lane n, 16 contiguous halfs at k0 + 16*(lane>=16).
// ---------------------------------------------------------------------------
template<int NT>
__device__ __forceinline__ void wmma_kloop(
    const unsigned short (*sA)[264], const unsigned short* __restrict__ Bw,
    int wave, int lane, v8f acc[NT])
{
  const int mlane = lane & 15;
  const int ka = (lane >> 4) * 16;
  const int kb = (lane >> 4) * 16;
  #pragma unroll
  for (int k0 = 0; k0 < 256; k0 += 32) {
    BfPack a;
    a.u[0] = *(const uint4*)&sA[mlane][k0 + ka];
    a.u[1] = *(const uint4*)&sA[mlane][k0 + ka + 8];
    #pragma unroll
    for (int t = 0; t < NT; ++t) {
      const int gn = (wave * NT + t) * 16 + mlane;
      const unsigned short* bp = Bw + ((size_t)gn << 8) + k0 + kb;
      BfPack b;
      b.u[0] = *(const uint4*)bp;
      b.u[1] = *(const uint4*)(bp + 8);
      acc[t] = __builtin_amdgcn_wmma_f32_16x16x32_bf16(
          false, a.v, false, b.v, (short)0, acc[t], false, false);
    }
  }
}

// ---------------------------------------------------------------------------
// GEMM with row-major fp32 A [M][256]: C[M][N] = A @ Bw^T + bias.  N = NT*128.
// One block = 8 waves = 16 M-rows x full N. Used for gemm1 (NT=3) & gemm2 (NT=2).
// ---------------------------------------------------------------------------
template<int NT>
__global__ __launch_bounds__(256) void gemm_rowA(
    const float* __restrict__ A, const unsigned short* __restrict__ Bw,
    const float* __restrict__ bias, float* __restrict__ C)
{
  constexpr int N = NT * 128;
  __shared__ __align__(16) unsigned short sA[16][264];
  const int tid = threadIdx.x;
  const int wave = tid >> 5, lane = tid & 31;
  const int m0 = blockIdx.x * 16;
  {
    const int row = tid >> 4;
    const int kc  = (tid & 15) << 4;
    const float* src = A + (size_t)(m0 + row) * 256 + kc;
    unsigned int pk[8];
    #pragma unroll
    for (int i = 0; i < 8; ++i)
      pk[i] = (unsigned int)f2bf(src[2 * i]) | ((unsigned int)f2bf(src[2 * i + 1]) << 16);
    uint4* d = (uint4*)&sA[row][kc];
    d[0] = make_uint4(pk[0], pk[1], pk[2], pk[3]);
    d[1] = make_uint4(pk[4], pk[5], pk[6], pk[7]);
  }
  __syncthreads();

  v8f acc[NT] = {};
  wmma_kloop<NT>(sA, Bw, wave, lane, acc);

  const int mlane = lane & 15;
  const int r0 = m0 + ((lane >> 4) << 3);
  #pragma unroll
  for (int t = 0; t < NT; ++t) {
    const int gn = (wave * NT + t) * 16 + mlane;
    const float bv = bias[gn];
    float* cp = C + (size_t)r0 * N + gn;
    #pragma unroll
    for (int v = 0; v < 8; ++v)
      cp[(size_t)v * N] = acc[t][v] + bv;
  }
}

// ---------------------------------------------------------------------------
// Value projection per level: feat is channel-major [B][256][HW]; stage a
// transposed bf16 tile [pos][k] in LDS, then same WMMA loop (M=pos, N=outchan).
// Output position-major: val[b][lvl_off+pos][256] (head channels contiguous).
// ---------------------------------------------------------------------------
__global__ __launch_bounds__(256) void gemm_val_kernel(
    const float* __restrict__ feat, const unsigned short* __restrict__ Wv,
    const float* __restrict__ bias, float* __restrict__ Vout,
    int HW, size_t lvl_off)
{
  __shared__ __align__(16) unsigned short sA[16][264];
  const int tid = threadIdx.x;
  const int wave = tid >> 5, lane = tid & 31;
  const int b = blockIdx.y;
  const int pos0 = blockIdx.x * 16;
  {
    const float* src = feat + ((size_t)b * 256 + tid) * (size_t)HW + pos0;
    const int nv = (HW - pos0 < 16) ? (HW - pos0) : 16;
    #pragma unroll
    for (int j = 0; j < 16; ++j) {
      const float f = (j < nv) ? src[j] : 0.f;
      sA[j][tid] = f2bf(f);                 // transpose while staging
    }
  }
  __syncthreads();

  v8f acc[2] = {};
  wmma_kloop<2>(sA, Wv, wave, lane, acc);

  const int mlane = lane & 15;
  const int pr = (lane >> 4) << 3;
  float* base = Vout + ((size_t)b * TOT_HW + lvl_off) * 256;
  #pragma unroll
  for (int t = 0; t < 2; ++t) {
    const int gn = (wave * 2 + t) * 16 + mlane;
    const float bv = bias[gn];
    #pragma unroll
    for (int v = 0; v < 8; ++v) {
      const int pos = pos0 + pr + v;
      if (pos < HW) base[(size_t)pos * 256 + gn] = acc[t][v] + bv;
    }
  }
}

// ---------------------------------------------------------------------------
// Sampler: one wave per (b,q,head); lane = head channel (hd==warpSize==32).
// tanh-bounded offsets + softmax over 16 (level,point) logits computed inline;
// align_corners bilinear simplifies to gx = ref.x*(W-1) + 0.1*2^l*tanh(offx).
// Corner reads are 32 contiguous floats (128 B) per corner -> coalesced.
// ---------------------------------------------------------------------------
__global__ __launch_bounds__(256) void sampler_kernel(
    const float* __restrict__ lin, const float* __restrict__ refp,
    const float* __restrict__ val, float* __restrict__ accum)
{
  const int lane = threadIdx.x & 31;
  const int gw = (blockIdx.x * 256 + threadIdx.x) >> 5;
  const int h  = gw & 7;
  const int bq = gw >> 3;
  if (bq >= B_ * Q_) return;
  const int b = bq >> 13;                    // Q_ = 8192
  const float* L = lin + (size_t)bq * 384;
  const float rx = refp[2 * bq], ry = refp[2 * bq + 1];

  float aw[16];
  float mx = -3.4e38f;
  #pragma unroll
  for (int j = 0; j < 16; ++j) { aw[j] = L[256 + h * 16 + j]; mx = fmaxf(mx, aw[j]); }
  float s = 0.f;
  #pragma unroll
  for (int j = 0; j < 16; ++j) { aw[j] = __expf(aw[j] - mx); s += aw[j]; }
  const float inv = 1.f / s;

  const int    WD[4] = {100, 50, 25, 13};
  const size_t LO[4] = {0, 10000, 12500, 13125};
  float acc = 0.f;
  #pragma unroll
  for (int l = 0; l < 4; ++l) {
    const int   Wd = WD[l];
    const float scale = 0.1f * (float)(1 << l);
    const float bx = rx * (float)(Wd - 1);
    const float by = ry * (float)(Wd - 1);
    const float* vb = val + ((size_t)b * TOT_HW + LO[l]) * 256 + h * 32 + lane;
    #pragma unroll
    for (int p = 0; p < 4; ++p) {
      const int oc = ((h * 4 + l) * 4 + p) * 2;
      const float gx = bx + scale * tanhf(L[oc]);
      const float gy = by + scale * tanhf(L[oc + 1]);
      const float xf = floorf(gx), yf = floorf(gy);
      const float fx = gx - xf, fy = gy - yf;
      const int x0 = (int)xf, y0 = (int)yf, x1 = x0 + 1, y1 = y0 + 1;
      const float m_x0 = (x0 >= 0 && x0 < Wd) ? 1.f : 0.f;
      const float m_x1 = (x1 >= 0 && x1 < Wd) ? 1.f : 0.f;
      const float m_y0 = (y0 >= 0 && y0 < Wd) ? 1.f : 0.f;
      const float m_y1 = (y1 >= 0 && y1 < Wd) ? 1.f : 0.f;
      const int cx0 = min(max(x0, 0), Wd - 1);
      const int cx1 = min(max(x1, 0), Wd - 1);
      const int cy0 = min(max(y0, 0), Wd - 1);
      const int cy1 = min(max(y1, 0), Wd - 1);
      const float s00 = vb[(size_t)(cy0 * Wd + cx0) * 256];
      const float s01 = vb[(size_t)(cy0 * Wd + cx1) * 256];
      const float s10 = vb[(size_t)(cy1 * Wd + cx0) * 256];
      const float s11 = vb[(size_t)(cy1 * Wd + cx1) * 256];
      const float a = aw[l * 4 + p];
      acc += a * ((1.f - fy) * ((1.f - fx) * m_x0 * m_y0 * s00 + fx * m_x1 * m_y0 * s01)
                +        fy  * ((1.f - fx) * m_x0 * m_y1 * s10 + fx * m_x1 * m_y1 * s11));
    }
  }
  accum[(size_t)bq * 256 + h * 32 + lane] = acc * inv;
}

// ---------------------------------------------------------------------------
// Workspace carve (bytes):
//   wcat 196608 | wval 524288 | wout 131072 | bias_cat 1536
//   lin 50331648 | val 54452224 | accum 33554432   (total ~139.2 MB)
// ---------------------------------------------------------------------------
extern "C" void kernel_launch(void* const* d_in, const int* in_sizes, int n_in,
                              void* d_out, int out_size, void* d_ws, size_t ws_size,
                              hipStream_t stream)
{
  const float* query  = (const float*)d_in[0];
  const float* refp   = (const float*)d_in[1];
  const float* feat[4] = {(const float*)d_in[2], (const float*)d_in[3],
                          (const float*)d_in[4], (const float*)d_in[5]};
  const float* w_off  = (const float*)d_in[6];
  const float* b_off  = (const float*)d_in[7];
  const float* w_attn = (const float*)d_in[8];
  const float* b_attn = (const float*)d_in[9];
  const float* w_val  = (const float*)d_in[10];
  const float* b_val  = (const float*)d_in[11];
  const float* w_out  = (const float*)d_in[12];
  const float* b_out  = (const float*)d_in[13];

  char* ws = (char*)d_ws;
  unsigned short* wcat     = (unsigned short*)(ws + 0);
  unsigned short* wval     = (unsigned short*)(ws + 196608);
  unsigned short* wout     = (unsigned short*)(ws + 720896);
  float*          bias_cat = (float*)(ws + 851968);
  float*          lin      = (float*)(ws + 853504);
  float*          valb     = (float*)(ws + 51185152);
  float*          accum    = (float*)(ws + 105637376);

  prep_kernel<<<1024, 256, 0, stream>>>(w_off, b_off, w_attn, b_attn, w_val, w_out,
                                        wcat, wval, wout, bias_cat);

  gemm_rowA<3><<<(B_ * Q_) / 16, 256, 0, stream>>>(query, wcat, bias_cat, lin);

  const int    HWs[4]  = {10000, 2500, 625, 169};
  const size_t offs[4] = {0, 10000, 12500, 13125};
  for (int l = 0; l < 4; ++l) {
    dim3 g((HWs[l] + 15) / 16, B_);
    gemm_val_kernel<<<g, 256, 0, stream>>>(feat[l], wval + (size_t)l * 65536,
                                           b_val + (size_t)l * 256, valb,
                                           HWs[l], offs[l]);
  }

  sampler_kernel<<<(B_ * Q_ * NHEAD) / 8, 256, 0, stream>>>(lin, refp, valb, accum);

  gemm_rowA<2><<<(B_ * Q_) / 16, 256, 0, stream>>>(accum, wout, b_out, (float*)d_out);

  (void)in_sizes; (void)n_in; (void)out_size; (void)ws_size;
}
